// LatentAttention_18605798326600
// MI455X (gfx1250) — matrix operand order repaired and checked by reading, not compiled, not run on hardware
//
#include <hip/hip_runtime.h>

// LatentAttention for MI455X (gfx1250, wave32, WMMA bf16 16x16x32).
//
// Pipeline:
//  1) Q = query@wq, K = key@wk, V = key@wv      (WMMA GEMM, fp32 in -> bf16 out)
//  2) softmax over per-head latent (rows of 128) on Q, in place
//  3) column stats over time on K: colmax[b,l], ksum[b,l] = sum_t exp(K-max)
//  4) per (b,h): Kv[128,128] = (exp(K-max))^T @ V * (1/ksum)  (WMMA, exp fused)
//  5) per (b,h): y = Qs @ Kv                     (WMMA bf16, async-LDS A tiles)
//  6) out = y @ out_proj                         (WMMA, async-LDS A tiles, fp32 out)

typedef __attribute__((ext_vector_type(16))) __bf16 v16bf;
typedef __attribute__((ext_vector_type(8)))  float  v8f;

union Frag { v16bf v; uint4 q[2]; };

__device__ __forceinline__ unsigned short f2bf(float f) {
  __bf16 h = (__bf16)f;                       // native v_cvt, RNE
  return __builtin_bit_cast(unsigned short, h);
}
__device__ __forceinline__ float bf2f(unsigned short s) {
  return __uint_as_float(((unsigned int)s) << 16);
}
__device__ __forceinline__ unsigned int pack2(float a, float b) {
  return (unsigned int)f2bf(a) | ((unsigned int)f2bf(b) << 16);
}
__device__ __forceinline__ void storeC(float* p, float v)          { *p = v; }
__device__ __forceinline__ void storeC(unsigned short* p, float v) { *p = f2bf(v); }
__device__ __forceinline__ v8f vzero() {
  v8f v = {0.f, 0.f, 0.f, 0.f, 0.f, 0.f, 0.f, 0.f};
  return v;
}

// ---- staging helpers -------------------------------------------------------
// 8 contiguous fp32 -> 8 bf16 in LDS as one b128 store.
__device__ __forceinline__ void stage8(const float* s, unsigned short* d) {
  float4 a = *(const float4*)s, b = *(const float4*)(s + 4);
  uint4 o;
  o.x = pack2(a.x, a.y); o.y = pack2(a.z, a.w);
  o.z = pack2(b.x, b.y); o.w = pack2(b.z, b.w);
  *(uint4*)d = o;
}

// 4x4 block transpose: rows k..k+3, cols n..n+3 of row-major [K x N] source
// -> LDS [n][k] (row stride BK), packed b64 stores.
template <int BK>
__device__ __forceinline__ void stageT4(const float* s, long long ld, unsigned short* d) {
  float4 r0 = *(const float4*)(s);
  float4 r1 = *(const float4*)(s + ld);
  float4 r2 = *(const float4*)(s + 2 * ld);
  float4 r3 = *(const float4*)(s + 3 * ld);
  uint2 o0 = { pack2(r0.x, r1.x), pack2(r2.x, r3.x) };
  uint2 o1 = { pack2(r0.y, r1.y), pack2(r2.y, r3.y) };
  uint2 o2 = { pack2(r0.z, r1.z), pack2(r2.z, r3.z) };
  uint2 o3 = { pack2(r0.w, r1.w), pack2(r2.w, r3.w) };
  *(uint2*)(d)          = o0;
  *(uint2*)(d + BK)     = o1;
  *(uint2*)(d + 2 * BK) = o2;
  *(uint2*)(d + 3 * BK) = o3;
}
template <int BK>
__device__ __forceinline__ void stageT4(const unsigned short* s, long long ld,
                                        unsigned short* d) {
  uint2 r0 = *(const uint2*)(s);
  uint2 r1 = *(const uint2*)(s + ld);
  uint2 r2 = *(const uint2*)(s + 2 * ld);
  uint2 r3 = *(const uint2*)(s + 3 * ld);
  // rX.x = {n, n+1}, rX.y = {n+2, n+3} as bf16 pairs; transpose to [n][k]
  uint2 o0 = { (r0.x & 0xffffu) | (r1.x << 16),  (r2.x & 0xffffu) | (r3.x << 16) };
  uint2 o1 = { (r0.x >> 16) | (r1.x & 0xffff0000u), (r2.x >> 16) | (r3.x & 0xffff0000u) };
  uint2 o2 = { (r0.y & 0xffffu) | (r1.y << 16),  (r2.y & 0xffffu) | (r3.y << 16) };
  uint2 o3 = { (r0.y >> 16) | (r1.y & 0xffff0000u), (r2.y >> 16) | (r3.y & 0xffff0000u) };
  *(uint2*)(d)          = o0;
  *(uint2*)(d + BK)     = o1;
  *(uint2*)(d + 2 * BK) = o2;
  *(uint2*)(d + 3 * BK) = o3;
}

// ---------------------------------------------------------------------------
// Tiled WMMA GEMM: C[M,N] = A[M,K] @ B[K,N]  (row-major, M,N % 128 == 0,
// K % 32 == 0). 256 threads = 8 waves in a 4(M) x 2(N) grid; each wave owns a
// 32x64 sub-tile = 2x4 WMMA accumulators. Double-buffered LDS so tile s+1
// loads overlap tile s WMMAs (one barrier per step). When A is already bf16
// the A tile is staged with GLOBAL_LOAD_ASYNC_TO_LDS_B128 (ASYNCcnt path).
// Batch offset: z = blockIdx.z; off = (z / Hdiv)*outStride + (z % Hdiv)*inStride.
// ---------------------------------------------------------------------------
template <typename TA, typename TB, typename TC>
__global__ __launch_bounds__(256) void gemm_wmma(
    const TA* __restrict__ A, int lda, long long aOut, long long aIn,
    const TB* __restrict__ B, int ldb, long long bOut, long long bIn,
    TC* __restrict__ C, int ldc, long long cOut, long long cIn,
    int Hdiv, int K)
{
  constexpr int BM = 128, BN = 128, BK = 32;
  constexpr bool ASYNC_A = __is_same(TA, unsigned short);
  __shared__ unsigned short sA[2][BM * BK];   // [m][k]
  __shared__ unsigned short sB[2][BN * BK];   // [n][k] (transposed tile)

  const long long zo = blockIdx.z / Hdiv;
  const long long zi = blockIdx.z % Hdiv;
  A += zo * aOut + zi * aIn;
  B += zo * bOut + zi * bIn;
  C += zo * cOut + zi * cIn;

  const int tid  = threadIdx.x;
  const int lane = tid & 31;
  const int wave = tid >> 5;
  const int wm   = wave & 3;    // M sub-tile: 32 rows each
  const int wn   = wave >> 2;   // N sub-tile: 64 cols each
  const int blockM = blockIdx.y * BM;
  const int blockN = blockIdx.x * BN;

  v8f acc[2][4];
#pragma unroll
  for (int i = 0; i < 2; ++i)
#pragma unroll
    for (int j = 0; j < 4; ++j) acc[i][j] = vzero();

  // staging coordinates
  const int aRow = tid >> 1;                 // 0..127
  const int aK   = (tid & 1) * 16;           // 0 or 16
  const int bN   = (tid & 31) * 4;           // 0..124 (n)
  const int bKq  = (tid >> 5) * 4;           // 0..28  (k)

  auto stage = [&](int k0, int buf) {
    { // A tile: 128 x 32 bf16 [m][k]; 16 elements (32 B) per thread
      const TA* src = A + (long long)(blockM + aRow) * lda + (k0 + aK);
      unsigned short* dst = &sA[buf][aRow * BK + aK];
      if constexpr (ASYNC_A) {
        // bf16 source: DMA straight to LDS, no VGPR round-trip.
        const unsigned int ldsa = (unsigned int)(unsigned long long)dst;
        asm volatile(
            "global_load_async_to_lds_b128 %0, %1, off\n\t"
            "global_load_async_to_lds_b128 %0, %1, off offset:16"
            :: "v"(ldsa), "v"(src) : "memory");
      } else {
        stage8(src, dst);
        stage8(src + 8, dst + 8);
      }
    }
    { // B tile: 32 x 128 -> bf16 [n][k]; 4x4 register transpose, b64 stores
      const TB* src = B + (long long)(k0 + bKq) * ldb + (blockN + bN);
      stageT4<BK>(src, (long long)ldb, &sB[buf][bN * BK + bKq]);
    }
  };

  const int nsteps = K / BK;
  stage(0, 0);
  if constexpr (ASYNC_A) asm volatile("s_wait_asynccnt 0x0" ::: "memory");
  __syncthreads();

  const int khalf = lane >> 4;
  const int fidx  = lane & 15;

  for (int s = 0; s < nsteps; ++s) {
    const int buf = s & 1;
    if (s + 1 < nsteps) stage((s + 1) * BK, buf ^ 1);   // overlap with compute

    Frag fa[2], fb[4];
#pragma unroll
    for (int i = 0; i < 2; ++i) {
      const unsigned short* base = &sA[buf][(wm * 32 + i * 16 + fidx) * BK];
      fa[i].q[0] = *(const uint4*)(base + khalf * 8);        // K {0-7 | 8-15}
      fa[i].q[1] = *(const uint4*)(base + 16 + khalf * 8);   // K {16-23 | 24-31}
    }
#pragma unroll
    for (int j = 0; j < 4; ++j) {
      const unsigned short* base = &sB[buf][(wn * 64 + j * 16 + fidx) * BK];
      fb[j].q[0] = *(const uint4*)(base + khalf * 8);        // K {0-7 | 8-15}
      fb[j].q[1] = *(const uint4*)(base + 16 + khalf * 8);   // K {16-23 | 24-31}
    }
#pragma unroll
    for (int i = 0; i < 2; ++i)
#pragma unroll
      for (int j = 0; j < 4; ++j)
        acc[i][j] = __builtin_amdgcn_wmma_f32_16x16x32_bf16(
            false, fa[i].v, false, fb[j].v, (short)0, acc[i][j], false, false);

    if constexpr (ASYNC_A) asm volatile("s_wait_asynccnt 0x0" ::: "memory");
    __syncthreads();
  }

  // C/D layout: VGPR r -> row (r | r+8) selected by lane>=16; col = lane&15
#pragma unroll
  for (int i = 0; i < 2; ++i) {
    const int row0 = blockM + wm * 32 + i * 16 + (lane >> 4) * 8;
#pragma unroll
    for (int j = 0; j < 4; ++j) {
      const int col = blockN + wn * 64 + j * 16 + (lane & 15);
#pragma unroll
      for (int r = 0; r < 8; ++r)
        storeC(&C[(long long)(row0 + r) * ldc + col], acc[i][j][r]);
    }
  }
}

// ---------------------------------------------------------------------------
// Softmax over rows of 128 bf16 values (per-head latent), in place.
// One wave per row, 4 elements per lane.
// ---------------------------------------------------------------------------
__global__ __launch_bounds__(256) void softmax128_kernel(unsigned short* Q, int nrows)
{
  const int row = blockIdx.x * 8 + (threadIdx.x >> 5);
  if (row >= nrows) return;
  const int lane = threadIdx.x & 31;
  unsigned short* p = Q + (long long)row * 128 + lane * 4;

  uint2 u = *(const uint2*)p;
  float x[4] = { bf2f((unsigned short)(u.x & 0xffff)), bf2f((unsigned short)(u.x >> 16)),
                 bf2f((unsigned short)(u.y & 0xffff)), bf2f((unsigned short)(u.y >> 16)) };

  float m = fmaxf(fmaxf(x[0], x[1]), fmaxf(x[2], x[3]));
#pragma unroll
  for (int off = 16; off > 0; off >>= 1) m = fmaxf(m, __shfl_xor(m, off, 32));

  float e[4], s = 0.f;
#pragma unroll
  for (int i = 0; i < 4; ++i) { e[i] = __expf(x[i] - m); s += e[i]; }
#pragma unroll
  for (int off = 16; off > 0; off >>= 1) s += __shfl_xor(s, off, 32);

  const float inv = 1.f / s;
  uint2 o = { pack2(e[0] * inv, e[1] * inv), pack2(e[2] * inv, e[3] * inv) };
  *(uint2*)p = o;
}

// ---------------------------------------------------------------------------
// Per-(b,l) stats over time: colmax and ksum = sum_t exp(K - colmax).
// One thread per (b,l); second pass hits L2 (K is 32 MB, L2 is 192 MB).
// ---------------------------------------------------------------------------
__global__ __launch_bounds__(256) void colstats_kernel(
    const unsigned short* __restrict__ K, float* __restrict__ colmax,
    float* __restrict__ ksum, int T, int L)
{
  const int idx = blockIdx.x * blockDim.x + threadIdx.x;   // b*L + l
  const int b = idx / L, l = idx % L;
  const unsigned short* p = K + (long long)b * T * L + l;
  float m = -3.0e38f;
  for (int t = 0; t < T; ++t) m = fmaxf(m, bf2f(p[(long long)t * L]));
  float s = 0.f;
  for (int t = 0; t < T; ++t) s += __expf(bf2f(p[(long long)t * L]) - m);
  colmax[idx] = m;
  ksum[idx]   = s;
}

// ---------------------------------------------------------------------------
// Per-(b,h): Kv[l,d] = (1/ksum[l]) * sum_t exp(K[t,l]-max[l]) * V[t,d]
// GEMM with M=N=128, inner dim T=4096; exp fused into the A-tile staging
// (which also does the [t][l] -> [l][t] transpose). Output bf16.
// ---------------------------------------------------------------------------
__global__ __launch_bounds__(256) void kv_kernel(
    const unsigned short* __restrict__ Kp, const unsigned short* __restrict__ Vp,
    const float* __restrict__ colmax, const float* __restrict__ ksum,
    unsigned short* __restrict__ Kv, int T)
{
  constexpr int BK = 32;
  __shared__ unsigned short sA[128 * BK];   // [l][t]
  __shared__ unsigned short sB[128 * BK];   // [d][t]
  __shared__ float smax[128];
  __shared__ float sinv[128];               // 1 / ksum

  const int z = blockIdx.x;          // b*8 + h
  const int b = z >> 3, h = z & 7;
  const int tid = threadIdx.x, lane = tid & 31, wave = tid >> 5;
  const int wm = wave & 3, wn = wave >> 2;
  const long long rowBase = (long long)b * T * 1024 + h * 128;

  if (tid < 128) {
    smax[tid] = colmax[b * 1024 + h * 128 + tid];
    sinv[tid] = 1.0f / ksum[b * 1024 + h * 128 + tid];
  }
  __syncthreads();

  v8f acc[2][4];
#pragma unroll
  for (int i = 0; i < 2; ++i)
#pragma unroll
    for (int j = 0; j < 4; ++j) acc[i][j] = vzero();

  const int cN = (tid & 31) * 4;      // l (A) / d (B) group of 4
  const int cK = (tid >> 5) * 4;      // t group of 4

  for (int t0 = 0; t0 < T; t0 += BK) {
    { // A tile: exp(K - max), 4x4 transpose [t][l] -> [l][t], packed stores
      const unsigned short* src = Kp + rowBase + (long long)(t0 + cK) * 1024 + cN;
      float e[4][4];
#pragma unroll
      for (int tt = 0; tt < 4; ++tt) {
        uint2 r = *(const uint2*)(src + (long long)tt * 1024);
        unsigned short sv[4] = { (unsigned short)(r.x & 0xffff), (unsigned short)(r.x >> 16),
                                 (unsigned short)(r.y & 0xffff), (unsigned short)(r.y >> 16) };
#pragma unroll
        for (int i = 0; i < 4; ++i) e[tt][i] = __expf(bf2f(sv[i]) - smax[cN + i]);
      }
      unsigned short* d = sA + cN * BK + cK;
#pragma unroll
      for (int i = 0; i < 4; ++i) {
        uint2 o = { pack2(e[0][i], e[1][i]), pack2(e[2][i], e[3][i]) };
        *(uint2*)(d + i * BK) = o;
      }
    }
    { // B tile: V 4x4 transpose [t][d] -> [d][t], raw bf16
      const unsigned short* src = Vp + rowBase + (long long)(t0 + cK) * 1024 + cN;
      stageT4<BK>(src, (long long)1024, sB + cN * BK + cK);
    }
    __syncthreads();

    const int khalf = lane >> 4;
    const int fidx  = lane & 15;
    Frag fa[2], fb[4];
#pragma unroll
    for (int i = 0; i < 2; ++i) {
      const unsigned short* base = sA + (wm * 32 + i * 16 + fidx) * BK;
      fa[i].q[0] = *(const uint4*)(base + khalf * 8);
      fa[i].q[1] = *(const uint4*)(base + 16 + khalf * 8);
    }
#pragma unroll
    for (int j = 0; j < 4; ++j) {
      const unsigned short* base = sB + (wn * 64 + j * 16 + fidx) * BK;
      fb[j].q[0] = *(const uint4*)(base + khalf * 8);
      fb[j].q[1] = *(const uint4*)(base + 16 + khalf * 8);
    }
#pragma unroll
    for (int i = 0; i < 2; ++i)
#pragma unroll
      for (int j = 0; j < 4; ++j)
        acc[i][j] = __builtin_amdgcn_wmma_f32_16x16x32_bf16(
            false, fa[i].v, false, fb[j].v, (short)0, acc[i][j], false, false);
    __syncthreads();
  }

#pragma unroll
  for (int i = 0; i < 2; ++i) {
    const int l0 = wm * 32 + i * 16 + (lane >> 4) * 8;
#pragma unroll
    for (int j = 0; j < 4; ++j) {
      const int col = wn * 64 + j * 16 + (lane & 15);
#pragma unroll
      for (int r = 0; r < 8; ++r) {
        const int l = l0 + r;
        Kv[(long long)z * 16384 + l * 128 + col] = f2bf(acc[i][j][r] * sinv[l]);
      }
    }
  }
}

// ---------------------------------------------------------------------------
// Host launcher
// ---------------------------------------------------------------------------
extern "C" void kernel_launch(void* const* d_in, const int* in_sizes, int n_in,
                              void* d_out, int out_size, void* d_ws, size_t ws_size,
                              hipStream_t stream)
{
  (void)in_sizes; (void)n_in; (void)out_size; (void)ws_size;
  constexpr int B = 4, T = 4096, D = 1024, LAT = 1024, H = 8;
  constexpr int M = B * T;                       // 16384
  constexpr long long QKV_BYTES = (long long)M * D * 2;  // 32 MiB each (bf16)

  const float* query = (const float*)d_in[0];
  const float* key   = (const float*)d_in[1];
  const float* wq    = (const float*)d_in[2];
  const float* wk    = (const float*)d_in[3];
  const float* wv    = (const float*)d_in[4];
  const float* wout  = (const float*)d_in[5];
  float* out = (float*)d_out;

  char* ws = (char*)d_ws;
  unsigned short* Q  = (unsigned short*)(ws);
  unsigned short* Kb = (unsigned short*)(ws + QKV_BYTES);
  unsigned short* V  = (unsigned short*)(ws + 2 * QKV_BYTES);
  unsigned short* Y  = Kb;   // K buffer is dead after kv_kernel -> reuse for y
  float* colmax = (float*)(ws + 3 * QKV_BYTES);
  float* ksum   = colmax + B * LAT;
  unsigned short* Kv = (unsigned short*)(ksum + B * LAT);  // B*H*128*128 bf16 = 1 MiB

  dim3 blk(256);

  // 1) Projections (fp32 -> bf16): grid (N/128, M/128)
  gemm_wmma<float, float, unsigned short><<<dim3(LAT / 128, M / 128, 1), blk, 0, stream>>>(
      query, D, 0, 0, wq, LAT, 0, 0, Q, LAT, 0, 0, 1, D);
  gemm_wmma<float, float, unsigned short><<<dim3(LAT / 128, M / 128, 1), blk, 0, stream>>>(
      key, D, 0, 0, wk, LAT, 0, 0, Kb, LAT, 0, 0, 1, D);
  gemm_wmma<float, float, unsigned short><<<dim3(D / 128, M / 128, 1), blk, 0, stream>>>(
      key, D, 0, 0, wv, D, 0, 0, V, D, 0, 0, 1, D);

  // 2) Softmax over per-head latent (rows of 128) on Q
  softmax128_kernel<<<dim3((M * H) / 8), blk, 0, stream>>>(Q, M * H);

  // 3) Column stats on K over time
  colstats_kernel<<<dim3((B * LAT) / 256), blk, 0, stream>>>(Kb, colmax, ksum, T, LAT);

  // 4) Kv per (b,h)
  kv_kernel<<<dim3(B * H), blk, 0, stream>>>(Kb, V, colmax, ksum, Kv, T);

  // 5) y = Qs @ Kv, batched over z = b*H + h (Hdiv = H); async A staging
  gemm_wmma<unsigned short, unsigned short, unsigned short>
      <<<dim3(1, T / 128, B * H), blk, 0, stream>>>(
      Q, LAT, (long long)T * LAT, 128,
      Kv, 128, (long long)H * 128 * 128, (long long)128 * 128,
      Y, D, (long long)T * D, 128,
      H, 128);

  // 6) out = y @ out_proj (bf16 x fp32 -> fp32 into d_out); async A staging
  gemm_wmma<unsigned short, float, float><<<dim3(D / 128, M / 128, 1), blk, 0, stream>>>(
      Y, D, 0, 0, wout, D, 0, 0, out, D, 0, 0, 1, D);
}
